// AttentionCTS_34419867910991
// MI455X (gfx1250) — matrix operand order repaired and checked by reading, not compiled, not run on hardware
//
#include <hip/hip_runtime.h>

// Problem constants: B=4, DIM=64, HEADS=4, T=10, H=64, W=64
#define NB    4
#define CDIM  64
#define NHEAD 4
#define TT    10
#define HWSZ  4096              // H*W
#define THW   40960             // T*H*W
#define NPOS  163840            // B*T*H*W
#define NTOK  160               // cph*T = 16*10 tokens per (b,head)
#define BH    16                // B*HEADS

typedef _Float16 v8h  __attribute__((ext_vector_type(8)));
typedef _Float16 v16h __attribute__((ext_vector_type(16)));
typedef float    v8f  __attribute__((ext_vector_type(8)));
typedef int      v4i  __attribute__((ext_vector_type(4)));

#if defined(__has_builtin)
#if __has_builtin(__builtin_amdgcn_global_load_async_to_lds_b128) && \
    __has_builtin(__builtin_amdgcn_s_wait_asynccnt)
#define ASYNC_LDS 1
#endif
#endif

// ---- WMMA helpers (CDNA5 v_wmma_f32_16x16x32_f16, wave32) ------------------
// A 16x32 f16 fragment: lane row = lane&15; lanes<16 hold K {0..15},
// lanes>=16 hold K {16..31}  -> two contiguous v8h loads.
__device__ __forceinline__ v16h load_afrag(const _Float16* p) {
    v8h lo = *(const v8h*)(p);
    v8h hi = *(const v8h*)(p + 8);
    v16h r;
#pragma unroll
    for (int i = 0; i < 8; ++i) { r[i] = lo[i]; r[i + 8] = hi[i]; }
    return r;
}
// B 32x16 f16 fragment: lane col = lane&15; K = (lane>>4)*16 + e -> one v16h.
__device__ __forceinline__ v16h load_bfrag(const _Float16* p) {
    return *(const v16h*)(p);
}
__device__ __forceinline__ v8f wmma32f16(v16h a, v16h b, v8f c) {
    return __builtin_amdgcn_wmma_f32_16x16x32_f16(false, a, false, b,
                                                  (short)0, c, false, false);
}

// ---- 0a: transpose x (b,c,s) fp32 -> xT[pos][c] f16 ------------------------
__global__ __launch_bounds__(256) void k_pack_x(const float* __restrict__ x,
                                                _Float16* __restrict__ xT) {
    int pos = blockIdx.x * 256 + threadIdx.x;          // 163840
    int b = pos / THW, s = pos % THW;
    const float* src = x + (size_t)b * CDIM * THW + s;
    _Float16* dst = xT + (size_t)pos * CDIM;
#pragma unroll 8
    for (int c = 0; c < CDIM; ++c) dst[c] = (_Float16)src[(size_t)c * THW];
}

// ---- 0b: weights fp32 -> f16 ----------------------------------------------
__global__ __launch_bounds__(256) void k_pack_w(const float* __restrict__ w_qkv,
                                                const float* __restrict__ w_out,
                                                _Float16* __restrict__ wqkvh,
                                                _Float16* __restrict__ wouth) {
    int i = blockIdx.x * 256 + threadIdx.x;            // 16384
    if (i < 192 * 64) wqkvh[i] = (_Float16)w_qkv[i];
    else              wouth[i - 192 * 64] = (_Float16)w_out[i - 192 * 64];
}

// ---- 1: QKV projection GEMM: D[o][pos] = sum_c wqkv[o][c] * xT[pos][c] -----
__global__ __launch_bounds__(256) void k_qkv_proj(const _Float16* __restrict__ xT,
                                                  const _Float16* __restrict__ wqkvh,
                                                  float* __restrict__ qkv) {
    int wave = blockIdx.x * 8 + (threadIdx.x >> 5);    // 10240 pos-tiles
    int lane = threadIdx.x & 31, hi = lane >> 4, lid = lane & 15;
    int pos = wave * 16 + lid;
    const _Float16* xrow = xT + (size_t)pos * CDIM;
    v16h b0 = load_bfrag(xrow + 0  + hi * 16);
    v16h b1 = load_bfrag(xrow + 32 + hi * 16);
    int b = pos / THW, s = pos % THW;                  // tile never crosses b
#pragma unroll 3
    for (int ot = 0; ot < 12; ++ot) {
        const _Float16* wrow = wqkvh + (size_t)(ot * 16 + lid) * CDIM;
        v16h a0 = load_afrag(wrow + 0  + hi * 16);
        v16h a1 = load_afrag(wrow + 32 + hi * 16);
        v8f acc = {};
        acc = wmma32f16(a0, b0, acc);
        acc = wmma32f16(a1, b1, acc);
#pragma unroll
        for (int r = 0; r < 8; ++r) {
            int o = ot * 16 + r + hi * 8;
            qkv[(size_t)(b * 192 + o) * THW + s] = acc[r];
        }
    }
}

// ---- 2: depthwise 3x3 conv over (h,w), 192 channels ------------------------
__global__ __launch_bounds__(256) void k_dwconv(const float* __restrict__ in,
                                                const float* __restrict__ w_dw,
                                                float* __restrict__ out) {
    int idx = blockIdx.x * 256 + threadIdx.x;          // 31457280
    int w = idx & 63, h = (idx >> 6) & 63;
    int c = (idx / THW) % 192;
    const float* wk = w_dw + c * 9;
    float acc = 0.f;
#pragma unroll
    for (int dh = -1; dh <= 1; ++dh) {
        int hh = h + dh;
        if (hh < 0 || hh > 63) continue;
#pragma unroll
        for (int dw = -1; dw <= 1; ++dw) {
            int ww = w + dw;
            if (ww < 0 || ww > 63) continue;
            acc += in[idx + dh * 64 + dw] * wk[(dh + 1) * 3 + (dw + 1)];
        }
    }
    out[idx] = acc;
}

// ---- 3: temporal mix: out[s] = b_t[s] + sum_t in[t]*w_t[s][t] --------------
__global__ __launch_bounds__(256) void k_temporal(const float* __restrict__ in,
                                                  const float* __restrict__ w_t,
                                                  const float* __restrict__ b_t,
                                                  float* __restrict__ out) {
    int idx = blockIdx.x * 256 + threadIdx.x;          // 3145728 = B*192*HW
    int hw = idx & (HWSZ - 1);
    int bc = idx >> 12;                                // b*192 + c
    size_t base = (size_t)bc * THW + hw;
    float v[TT];
#pragma unroll
    for (int t = 0; t < TT; ++t) v[t] = in[base + (size_t)t * HWSZ];
#pragma unroll
    for (int s = 0; s < TT; ++s) {
        float acc = b_t[s];
#pragma unroll
        for (int t = 0; t < TT; ++t) acc += v[t] * w_t[s * TT + t];
        out[base + (size_t)s * HWSZ] = acc;
    }
}

// ---- 4: L2-normalize q,k rows; pack q,k f16 row-major, v f16 transposed ----
__global__ __launch_bounds__(256) void k_normalize(const float* __restrict__ qkv,
                                                   _Float16* __restrict__ qh,
                                                   _Float16* __restrict__ kh,
                                                   _Float16* __restrict__ vT) {
    __shared__ float rq[256], rk[256];
    int row = blockIdx.x;                              // 2560 = BH*NTOK
    int tid = threadIdx.x;
    int bh = row / NTOK, n = row % NTOK;
    int b = bh >> 2, head = bh & 3;
    int c = head * 16 + n / TT, t = n % TT;
    size_t base = ((size_t)(b * 192 + c) * TT + t) * HWSZ;
    const float* q = qkv + base;
    const float* k = qkv + base + (size_t)64 * THW;
    const float* v = qkv + base + (size_t)128 * THW;
    float sq = 0.f, sk = 0.f;
    for (int d = tid; d < HWSZ; d += 256) {
        float a = q[d], bb = k[d];
        sq += a * a; sk += bb * bb;
    }
    rq[tid] = sq; rk[tid] = sk;
    __syncthreads();
    for (int st = 128; st > 0; st >>= 1) {
        if (tid < st) { rq[tid] += rq[tid + st]; rk[tid] += rk[tid + st]; }
        __syncthreads();
    }
    float invq = 1.0f / fmaxf(sqrtf(rq[0]), 1e-12f);
    float invk = 1.0f / fmaxf(sqrtf(rk[0]), 1e-12f);
    _Float16* qd = qh + (size_t)row * HWSZ;
    _Float16* kd = kh + (size_t)row * HWSZ;
    for (int d = tid; d < HWSZ; d += 256) {
        qd[d] = (_Float16)(q[d] * invq);
        kd[d] = (_Float16)(k[d] * invk);
        vT[((size_t)bh * HWSZ + d) * NTOK + n] = (_Float16)v[d];
    }
}

// ---- 5: S = Qn * Kn^T * temperature; M=N=160, K=4096 per (b,h) -------------
// One workgroup (10 waves) per (bh, mt): wave w owns key-tile nt=w.
// The shared 16x4096 Q tile is staged through LDS in 16x1024 chunks using
// gfx1250 async global->LDS copies (Q traffic cut 10x); A-fragments then come
// from LDS (ds_load_b128), B-fragments stream from global.
#define KCHUNK 1024
__global__ __launch_bounds__(320) void k_scores(const _Float16* __restrict__ qh,
                                                const _Float16* __restrict__ kh,
                                                const float* __restrict__ temperature,
                                                float* __restrict__ attnf) {
    __shared__ _Float16 qtile[16 * KCHUNK];            // 32 KB
    int tid = threadIdx.x;
    int bh = blockIdx.x / 10, mt = blockIdx.x % 10;    // grid = 160
    int nt = tid >> 5;                                 // wave id = key tile
    int lane = tid & 31, hi = lane >> 4, lid = lane & 15;
    const _Float16* qbase = qh + (size_t)(bh * NTOK + mt * 16) * HWSZ;
    const _Float16* krow  = kh + (size_t)(bh * NTOK + nt * 16 + lid) * HWSZ;
    v8f acc = {};
    for (int ch = 0; ch < HWSZ / KCHUNK; ++ch) {
        // ---- stage Q chunk: 16 rows x 1024 halves = 2048 x b128 ----
        for (int i = tid; i < 16 * KCHUNK / 8; i += 320) {
            int row = i >> 7, off = i & 127;           // 128 b128 per row
            const _Float16* g = qbase + (size_t)row * HWSZ + ch * KCHUNK + off * 8;
#if ASYNC_LDS
            __builtin_amdgcn_global_load_async_to_lds_b128(
                (v4i*)g, (v4i*)&qtile[(size_t)i * 8], 0, 0);
#else
            *(v8h*)&qtile[(size_t)i * 8] = *(const v8h*)g;
#endif
        }
#if ASYNC_LDS
        __builtin_amdgcn_s_wait_asynccnt(0);
#endif
        __syncthreads();
        __builtin_prefetch(krow + (ch + 1) * KCHUNK, 0, 0);  // speculative
        const _Float16* arow = &qtile[lid * KCHUNK];
        const _Float16* brow = krow + ch * KCHUNK;
#pragma unroll 4
        for (int k0 = 0; k0 < KCHUNK; k0 += 32) {
            v16h a = load_afrag(arow + k0 + hi * 16);  // ds_load_b128 x2
            v16h b = load_bfrag(brow + k0 + hi * 16);  // global b128x2
            acc = wmma32f16(a, b, acc);
        }
        __syncthreads();                               // qtile reuse barrier
    }
    float temp = temperature[bh & 3];
    size_t obase = (size_t)bh * NTOK * NTOK + nt * 16 + lid;
#pragma unroll
    for (int r = 0; r < 8; ++r) {
        int m = mt * 16 + r + hi * 8;
        attnf[obase + (size_t)m * NTOK] = acc[r] * temp;
    }
}

// ---- 6: softmax over keys (160), one wave per row --------------------------
__global__ __launch_bounds__(256) void k_softmax(const float* __restrict__ attnf,
                                                 _Float16* __restrict__ attnh) {
    int row = blockIdx.x * 8 + (threadIdx.x >> 5);     // 2560 rows
    int lane = threadIdx.x & 31;
    const float* src = attnf + (size_t)row * NTOK;
    float v[5], mx = -3.0e38f;
#pragma unroll
    for (int i = 0; i < 5; ++i) { v[i] = src[lane + 32 * i]; mx = fmaxf(mx, v[i]); }
#pragma unroll
    for (int off = 16; off > 0; off >>= 1) mx = fmaxf(mx, __shfl_xor(mx, off, 32));
    float sum = 0.f;
#pragma unroll
    for (int i = 0; i < 5; ++i) { v[i] = __expf(v[i] - mx); sum += v[i]; }
#pragma unroll
    for (int off = 16; off > 0; off >>= 1) sum += __shfl_xor(sum, off, 32);
    float inv = 1.0f / sum;
    _Float16* dst = attnh + (size_t)row * NTOK;
#pragma unroll
    for (int i = 0; i < 5; ++i) dst[lane + 32 * i] = (_Float16)(v[i] * inv);
}

// ---- 7: out = attn(160x160) * V(160x4096); write outT[pos][c] f16 ----------
__global__ __launch_bounds__(256) void k_attn_v(const _Float16* __restrict__ attnh,
                                                const _Float16* __restrict__ vT,
                                                _Float16* __restrict__ outT) {
    int wave = blockIdx.x * 8 + (threadIdx.x >> 5);    // 40960 tiles
    int bh = wave / 2560;
    int rem = wave % 2560;
    int mt = rem / 256, nt = rem % 256;
    int lane = threadIdx.x & 31, hi = lane >> 4, lid = lane & 15;
    const _Float16* arow = attnh + (size_t)(bh * NTOK + mt * 16 + lid) * NTOK;
    const _Float16* brow = vT + ((size_t)bh * HWSZ + nt * 16 + lid) * NTOK;
    v8f acc = {};
#pragma unroll
    for (int k0 = 0; k0 < NTOK; k0 += 32) {
        v16h a = load_afrag(arow + k0 + hi * 16);
        v16h b = load_bfrag(brow + k0 + hi * 16);
        acc = wmma32f16(a, b, acc);
    }
    int b = bh >> 2, head = bh & 3;
    int hw = nt * 16 + lid;
#pragma unroll
    for (int r = 0; r < 8; ++r) {
        int nq = mt * 16 + r + hi * 8;
        int c = head * 16 + nq / TT, t = nq % TT;
        size_t p = (size_t)b * THW + (size_t)t * HWSZ + hw;
        outT[p * CDIM + c] = (_Float16)acc[r];
    }
}

// ---- 8: output projection: D[o][pos] = sum_c wout[o][c]*outT[pos][c] -------
__global__ __launch_bounds__(256) void k_out_proj(const _Float16* __restrict__ outT,
                                                  const _Float16* __restrict__ wouth,
                                                  float* __restrict__ y) {
    int wave = blockIdx.x * 8 + (threadIdx.x >> 5);    // 10240 pos-tiles
    int lane = threadIdx.x & 31, hi = lane >> 4, lid = lane & 15;
    int pos = wave * 16 + lid;
    const _Float16* xrow = outT + (size_t)pos * CDIM;
    v16h b0 = load_bfrag(xrow + 0  + hi * 16);
    v16h b1 = load_bfrag(xrow + 32 + hi * 16);
    int b = pos / THW, s = pos % THW;
#pragma unroll
    for (int ot = 0; ot < 4; ++ot) {
        const _Float16* wrow = wouth + (size_t)(ot * 16 + lid) * CDIM;
        v16h a0 = load_afrag(wrow + 0  + hi * 16);
        v16h a1 = load_afrag(wrow + 32 + hi * 16);
        v8f acc = {};
        acc = wmma32f16(a0, b0, acc);
        acc = wmma32f16(a1, b1, acc);
#pragma unroll
        for (int r = 0; r < 8; ++r) {
            int o = ot * 16 + r + hi * 8;
            y[(size_t)(b * CDIM + o) * THW + s] = acc[r];
        }
    }
}

// ---------------------------------------------------------------------------
extern "C" void kernel_launch(void* const* d_in, const int* in_sizes, int n_in,
                              void* d_out, int out_size, void* d_ws, size_t ws_size,
                              hipStream_t stream) {
    (void)in_sizes; (void)n_in; (void)out_size; (void)ws_size;
    const float* x     = (const float*)d_in[0];
    const float* w_qkv = (const float*)d_in[1];
    const float* w_dw  = (const float*)d_in[2];
    const float* w_t   = (const float*)d_in[3];
    const float* b_t   = (const float*)d_in[4];
    const float* temp  = (const float*)d_in[5];
    const float* w_out = (const float*)d_in[6];
    float* y = (float*)d_out;

    char* ws = (char*)d_ws;
    const size_t REG = 125829120;                  // 192*THW*B*4 bytes
    float*    qkvA  = (float*)(ws);                // region A
    float*    qkvB  = (float*)(ws + REG);          // region B
    _Float16* xT    = (_Float16*)(ws + REG);       // aliases B (dead by k2)
    _Float16* wqkvh = (_Float16*)(ws + 2 * REG);
    _Float16* wouth = (_Float16*)(ws + 2 * REG + 24576);
    // After temporal mix (qkvC == region A), region B is recycled:
    _Float16* qh    = (_Float16*)(ws + REG);
    _Float16* kh    = (_Float16*)(ws + REG + 20971520);
    _Float16* vT    = (_Float16*)(ws + REG + 2 * 20971520);
    // After normalize, region A is recycled:
    float*    attnf = (float*)(ws);
    _Float16* attnh = (_Float16*)(ws + 1638400);
    _Float16* outT  = (_Float16*)(ws + 2457600);

    k_pack_x  <<<640,    256, 0, stream>>>(x, xT);
    k_pack_w  <<<64,     256, 0, stream>>>(w_qkv, w_out, wqkvh, wouth);
    k_qkv_proj<<<1280,   256, 0, stream>>>(xT, wqkvh, qkvA);
    k_dwconv  <<<122880, 256, 0, stream>>>(qkvA, w_dw, qkvB);
    k_temporal<<<12288,  256, 0, stream>>>(qkvB, w_t, b_t, qkvA);
    k_normalize<<<2560,  256, 0, stream>>>(qkvA, qh, kh, vT);
    k_scores  <<<160,    320, 0, stream>>>(qh, kh, temp, attnf);
    k_softmax <<<320,    256, 0, stream>>>(attnf, attnh);
    k_attn_v  <<<5120,   256, 0, stream>>>(attnh, vT, outT);
    k_out_proj<<<1280,   256, 0, stream>>>(outT, wouth, y);
}
